// IDRSAGE_79018808312060
// MI455X (gfx1250) — matrix operand order repaired and checked by reading, not compiled, not run on hardware
//
#include <hip/hip_runtime.h>
#include <hip/hip_bf16.h>

#define N_SRC_C 200000
#define N_DST_C 100000
#define NE_C    500000
#define NP_C    100000
#define NPQ_C   200000
#define HASH_SZ 262144   // 2^18 slots per tensor, load factor ~0.38
#define LDSK    144      // padded LDS row (f16 elems): 288B rows -> 32B aligned frags

typedef __attribute__((ext_vector_type(16))) _Float16 v16h;
typedef __attribute__((ext_vector_type(8)))  float    v8f;

// ---------------------------------------------------------------------------
// Generic 128x128-weight GEMM:
//   acc  = Cin ? Cin[rows] : broadcast(bias)
//   acc += A0@W0 (+ A1@W1)          (A converted to f16, f32 WMMA accumulate)
//   if idxA: A-operand rows are A0[idxA[r]] * A0[idxB[r]] (fused gather-mul)
//   out = do_relu ? relu(acc) : acc
// Block = 256 thr (8 waves); each wave owns a 16-row x 128-col tile via WMMA.
// ---------------------------------------------------------------------------
__global__ __launch_bounds__(256)
void gemm128_kernel(const float* __restrict__ A0, const float* __restrict__ W0,
                    const float* __restrict__ A1, const float* __restrict__ W1,
                    const float* __restrict__ bias, const float* __restrict__ Cin,
                    const int* __restrict__ idxA, const int* __restrict__ idxB,
                    float* __restrict__ out, int M, int do_relu)
{
    __shared__ __align__(32) _Float16 Wt[128 * LDSK];   // W^T, f16, padded rows
    const int tid   = threadIdx.x;
    const int wave  = tid >> 5;
    const int lane  = tid & 31;
    const int rowBase = blockIdx.x * 128 + wave * 16;
    const int arow  = lane & 15;      // A-fragment row within 16-row tile
    const int khalf = lane >> 4;      // 0 / 1
    const int col0  = lane & 15;      // C/D column within 16-col tile

    v8f c[8];
    if (Cin) {
#pragma unroll
        for (int nt = 0; nt < 8; ++nt)
#pragma unroll
            for (int r = 0; r < 8; ++r) {
                const int orow = rowBase + khalf * 8 + r;
                const int rs = orow < M ? orow : M - 1;
                c[nt][r] = Cin[(size_t)rs * 128 + nt * 16 + col0];
            }
    } else {
#pragma unroll
        for (int nt = 0; nt < 8; ++nt) {
            const float bv = bias[nt * 16 + col0];
#pragma unroll
            for (int r = 0; r < 8; ++r) c[nt][r] = bv;
        }
    }

    for (int pass = 0; pass < 2; ++pass) {
        const float* A = (pass == 0) ? A0 : A1;
        const float* W = (pass == 0) ? W0 : W1;
        if (!A) break;                          // uniform

        __syncthreads();                        // prior pass done reading Wt
        for (int idx = tid; idx < 128 * 128; idx += 256) {
            const int k = idx >> 7, n = idx & 127;
            Wt[n * LDSK + k] = (_Float16)W[idx];
        }
        __syncthreads();

        const int row  = rowBase + arow;
        const int rs   = row < M ? row : M - 1;   // clamp: garbage only hits unsaved rows
        const float* ar0;
        const float* ar1 = nullptr;
        if (idxA) {
            ar0 = A + (size_t)idxA[rs] * 128;
            ar1 = A + (size_t)idxB[rs] * 128;
        } else {
            ar0 = A + (size_t)rs * 128;
        }

#pragma unroll
        for (int kt = 0; kt < 4; ++kt) {
            const int kA = kt * 32 + khalf * 16;
            float4 f0 = *(const float4*)(ar0 + kA);
            float4 f1 = *(const float4*)(ar0 + kA + 4);
            float4 f2 = *(const float4*)(ar0 + kA + 8);
            float4 f3 = *(const float4*)(ar0 + kA + 12);
            if (ar1) {
                const float4 g0 = *(const float4*)(ar1 + kA);
                const float4 g1 = *(const float4*)(ar1 + kA + 4);
                const float4 g2 = *(const float4*)(ar1 + kA + 8);
                const float4 g3 = *(const float4*)(ar1 + kA + 12);
                f0.x *= g0.x; f0.y *= g0.y; f0.z *= g0.z; f0.w *= g0.w;
                f1.x *= g1.x; f1.y *= g1.y; f1.z *= g1.z; f1.w *= g1.w;
                f2.x *= g2.x; f2.y *= g2.y; f2.z *= g2.z; f2.w *= g2.w;
                f3.x *= g3.x; f3.y *= g3.y; f3.z *= g3.z; f3.w *= g3.w;
            }
            v16h a;
            a[0]=(_Float16)f0.x; a[1]=(_Float16)f0.y; a[2]=(_Float16)f0.z; a[3]=(_Float16)f0.w;
            a[4]=(_Float16)f1.x; a[5]=(_Float16)f1.y; a[6]=(_Float16)f1.z; a[7]=(_Float16)f1.w;
            a[8]=(_Float16)f2.x; a[9]=(_Float16)f2.y; a[10]=(_Float16)f2.z; a[11]=(_Float16)f2.w;
            a[12]=(_Float16)f3.x; a[13]=(_Float16)f3.y; a[14]=(_Float16)f3.z; a[15]=(_Float16)f3.w;

            const int kB = kt * 32 + khalf * 16;
            // double-buffered B fragments: overlap ds_load with v_wmma
            v16h bcur = *(const v16h*)&Wt[col0 * LDSK + kB];
#pragma unroll
            for (int nt = 0; nt < 8; ++nt) {
                v16h bnext = bcur;
                if (nt < 7) bnext = *(const v16h*)&Wt[((nt + 1) * 16 + col0) * LDSK + kB];
                c[nt] = __builtin_amdgcn_wmma_f32_16x16x32_f16(
                            false, a, false, bcur, (short)0, c[nt], false, false);
                bcur = bnext;
            }
        }
    }

#pragma unroll
    for (int nt = 0; nt < 8; ++nt) {
#pragma unroll
        for (int r = 0; r < 8; ++r) {
            const int orow = rowBase + khalf * 8 + r;
            if (orow < M) {
                float v = c[nt][r];
                if (do_relu) v = v > 0.0f ? v : 0.0f;
                out[(size_t)orow * 128 + nt * 16 + col0] = v;
            }
        }
    }
}

// ---------------------------------------------------------------------------
__global__ void zero_kernel(float* __restrict__ p, size_t n)
{
    size_t i = blockIdx.x * (size_t)blockDim.x + threadIdx.x;
    const size_t stride = (size_t)gridDim.x * blockDim.x;
    for (; i < n; i += stride) p[i] = 0.0f;
}

// segment-max of relu'd rows: values >= 0 -> uint bit pattern is order-preserving
__global__ void scatter_max_kernel(const float* __restrict__ m, const int* __restrict__ src,
                                   const int* __restrict__ dst, float* __restrict__ neigh)
{
    const int idx = blockIdx.x * blockDim.x + threadIdx.x;   // NE*32 threads
    if (idx >= NE_C * 32) return;
    const int e = idx >> 5, c4 = idx & 31;
    const int s = src[e], d = dst[e];
    const float4 v = *(const float4*)(m + (size_t)s * 128 + c4 * 4);
    unsigned int* o = (unsigned int*)(neigh + (size_t)d * 128 + c4 * 4);
    atomicMax(o + 0, __float_as_uint(v.x));
    atomicMax(o + 1, __float_as_uint(v.y));
    atomicMax(o + 2, __float_as_uint(v.z));
    atomicMax(o + 3, __float_as_uint(v.w));
}

__global__ void segsum_kernel(const float* __restrict__ x, const int* __restrict__ src,
                              const int* __restrict__ dst, float* __restrict__ sums,
                              float* __restrict__ deg)
{
    const int t = blockIdx.y;                                 // 0..2
    const int idx = blockIdx.x * blockDim.x + threadIdx.x;    // NE*32 threads
    if (idx >= NE_C * 32) return;
    const int e = idx >> 5, c4 = idx & 31;
    const int s = src[t * NE_C + e], d = dst[t * NE_C + e];
    const float4 v = *(const float4*)(x + (size_t)s * 128 + c4 * 4);
    float* o = sums + (size_t)t * N_DST_C * 128 + (size_t)d * 128 + c4 * 4;
    atomicAdd(o + 0, v.x);
    atomicAdd(o + 1, v.y);
    atomicAdd(o + 2, v.z);
    atomicAdd(o + 3, v.w);
    if (c4 == 0) atomicAdd(deg + t * N_DST_C + d, 1.0f);
}

__global__ void mean_kernel(float* __restrict__ sums, const float* __restrict__ deg)
{
    const int t = blockIdx.y;
    const int idx = blockIdx.x * blockDim.x + threadIdx.x;    // N_DST*32 threads
    if (idx >= N_DST_C * 32) return;
    const int d = idx >> 5, c4 = idx & 31;
    const float dg = fmaxf(deg[t * N_DST_C + d], 1.0f);
    float4* p = (float4*)(sums + (size_t)t * N_DST_C * 128 + (size_t)d * 128 + c4 * 4);
    float4 v = *p;
    v.x /= dg; v.y /= dg; v.z /= dg; v.w /= dg;
    *p = v;
}

// salted per-row 64-bit hash (wave32 xor-reduce) -> open-addressed set -> n_unique
__global__ void hash_count_kernel(const float* __restrict__ hiso, const float* __restrict__ hc,
                                  unsigned long long* __restrict__ tables,
                                  unsigned int* __restrict__ uniq)
{
    const int t = blockIdx.y;   // 0=h_iso, 1..3 = h_comm[t-1]
    const float* base = (t == 0) ? hiso : (hc + (size_t)(t - 1) * N_DST_C * 128);
    const int wave = threadIdx.x >> 5, lane = threadIdx.x & 31;
    const int row = blockIdx.x * 8 + wave;
    if (row >= N_DST_C) return;

    const unsigned int* p = (const unsigned int*)(base + (size_t)row * 128) + lane * 4;
    unsigned long long h = 1469598103934665603ULL ^ (unsigned long long)(lane * 0x9E3779B9u);
#pragma unroll
    for (int j = 0; j < 4; ++j) {
        h ^= (unsigned long long)p[j] + (unsigned long long)j * 0x100000001B3ULL;
        h *= 1099511628211ULL;
        h ^= h >> 29;
    }
    unsigned int lo = (unsigned int)h, hi = (unsigned int)(h >> 32);
#pragma unroll
    for (int off = 16; off > 0; off >>= 1) {
        lo ^= __shfl_xor(lo, off, 32);
        hi ^= __shfl_xor(hi, off, 32);
    }
    if (lane == 0) {
        unsigned long long hv = (((unsigned long long)hi << 32) | lo) | 1ULL;
        unsigned long long* tab = tables + (size_t)t * HASH_SZ;
        unsigned int slot = (unsigned int)(hv ^ (hv >> 21)) & (HASH_SZ - 1);
        for (int probe = 0; probe < HASH_SZ; ++probe) {
            const unsigned long long prev = atomicCAS(&tab[slot], 0ULL, hv);
            if (prev == 0ULL) { atomicAdd(&uniq[t], 1u); break; }
            if (prev == hv) break;
            slot = (slot + 1) & (HASH_SZ - 1);
        }
    }
}

// h[i] = max over 4 tensors of (i < n_unique_t - 1 ? tensor_t[i] : 0)
__global__ void combine_max_kernel(const float* __restrict__ hiso, const float* __restrict__ hc,
                                   const unsigned int* __restrict__ uniq, float* __restrict__ h)
{
    const int idx = blockIdx.x * blockDim.x + threadIdx.x;    // N_DST*32 threads
    if (idx >= N_DST_C * 32) return;
    const int row = idx >> 5, c4 = idx & 31;
    const int th0 = (int)uniq[0] - 1, th1 = (int)uniq[1] - 1;
    const int th2 = (int)uniq[2] - 1, th3 = (int)uniq[3] - 1;
    const size_t off = (size_t)row * 128 + c4 * 4;
    const float4 z4 = make_float4(0.f, 0.f, 0.f, 0.f);
    const float4 v0 = (row < th0) ? *(const float4*)(hiso + off) : z4;
    const float4 v1 = (row < th1) ? *(const float4*)(hc + off) : z4;
    const float4 v2 = (row < th2) ? *(const float4*)(hc + (size_t)N_DST_C * 128 + off) : z4;
    const float4 v3 = (row < th3) ? *(const float4*)(hc + (size_t)2 * N_DST_C * 128 + off) : z4;
    float4 r;
    r.x = fmaxf(fmaxf(v0.x, v1.x), fmaxf(v2.x, v3.x));
    r.y = fmaxf(fmaxf(v0.y, v1.y), fmaxf(v2.y, v3.y));
    r.z = fmaxf(fmaxf(v0.z, v1.z), fmaxf(v2.z, v3.z));
    r.w = fmaxf(fmaxf(v0.w, v1.w), fmaxf(v2.w, v3.w));
    *(float4*)(h + off) = r;
}

__global__ void final_dot_kernel(const float* __restrict__ z, const float* __restrict__ W3,
                                 const float* __restrict__ b3, float* __restrict__ out, int M)
{
    const int i = blockIdx.x * blockDim.x + threadIdx.x;
    if (i >= M) return;
    const float4* zr = (const float4*)(z + (size_t)i * 128);
    const float4* w  = (const float4*)W3;
    float s = 0.0f;
#pragma unroll
    for (int j = 0; j < 32; ++j) {
        const float4 a = zr[j], b = w[j];
        s += a.x * b.x + a.y * b.y + a.z * b.z + a.w * b.w;
    }
    out[i] = s + b3[0];
}

// ---------------------------------------------------------------------------
extern "C" void kernel_launch(void* const* d_in, const int* in_sizes, int n_in,
                              void* d_out, int out_size, void* d_ws, size_t ws_size,
                              hipStream_t stream)
{
    const float* x        = (const float*)d_in[0];
    const int*   src_iso  = (const int*)d_in[1];
    const int*   dst_iso  = (const int*)d_in[2];
    const int*   src_comm = (const int*)d_in[3];
    const int*   dst_comm = (const int*)d_in[4];
    const int*   pos_src  = (const int*)d_in[5];
    const int*   pos_dst  = (const int*)d_in[6];
    const int*   neg_src  = (const int*)d_in[7];
    const int*   neg_dst  = (const int*)d_in[8];
    const float* W_pool   = (const float*)d_in[9];
    const float* b_pool   = (const float*)d_in[10];
    const float* W_self_i = (const float*)d_in[11];
    const float* W_neigh_i= (const float*)d_in[12];
    const float* b_i      = (const float*)d_in[13];
    const float* W_self_c = (const float*)d_in[14];
    const float* W_neigh_c= (const float*)d_in[15];
    const float* b_c      = (const float*)d_in[16];
    const float* W1       = (const float*)d_in[17];
    const float* b1       = (const float*)d_in[18];
    const float* W2       = (const float*)d_in[19];
    const float* b2       = (const float*)d_in[20];
    const float* W3       = (const float*)d_in[21];
    const float* b3       = (const float*)d_in[22];

    // ---- workspace layout (floats), with lifetime-based reuse ----
    float* ws = (float*)d_ws;
    size_t o = 0;
    float* sum_c = ws + o; o += (size_t)3 * N_DST_C * 128;   // seg-sums -> means
    float* mbuf  = ws + o; o += (size_t)N_SRC_C * 128;       // m; later z1
    float* neigh = ws + o; o += (size_t)N_DST_C * 128;       // neigh_i (zero-init)
    float* hiso  = ws + o; o += (size_t)N_DST_C * 128;
    float* hc    = ws + o; o += (size_t)3 * N_DST_C * 128;   // later z2
    float* hbuf  = ws + o; o += (size_t)N_DST_C * 128;       // base_c; later h
    float* deg   = ws + o; o += (size_t)3 * N_DST_C;         // zero-init
    unsigned long long* tables = (unsigned long long*)(ws + o); o += (size_t)4 * HASH_SZ * 2;
    unsigned int* uniq = (unsigned int*)(ws + o); o += 4;
    float* z1 = mbuf;    // reuse after scatter-max
    float* z2 = hc;      // reuse after combine

    // ---- 0. zero scratch that must start at zero (every call: replay-safe) ----
    zero_kernel<<<2048, 256, 0, stream>>>(neigh, (size_t)N_DST_C * 128);
    zero_kernel<<<2048, 256, 0, stream>>>(sum_c, (size_t)3 * N_DST_C * 128);
    zero_kernel<<<2048, 256, 0, stream>>>(deg, (size_t)3 * N_DST_C + (size_t)4 * HASH_SZ * 2 + 4);

    // ---- 1. m = relu(x @ W_pool + b_pool) ----
    gemm128_kernel<<<(N_SRC_C + 127) / 128, 256, 0, stream>>>(
        x, W_pool, nullptr, nullptr, b_pool, nullptr, nullptr, nullptr, mbuf, N_SRC_C, 1);

    // ---- 2. neigh_i = segment_max(m[src_iso], dst_iso) (empty -> 0) ----
    scatter_max_kernel<<<(NE_C * 32 + 255) / 256, 256, 0, stream>>>(mbuf, src_iso, dst_iso, neigh);

    // ---- 3. community segment sums + degrees, then means ----
    segsum_kernel<<<dim3((NE_C * 32 + 255) / 256, 3), 256, 0, stream>>>(
        x, src_comm, dst_comm, sum_c, deg);
    mean_kernel<<<dim3((N_DST_C * 32 + 255) / 256, 3), 256, 0, stream>>>(sum_c, deg);

    // ---- 4. h_iso = x_dst@W_self_i + neigh@W_neigh_i + b_i ----
    gemm128_kernel<<<(N_DST_C + 127) / 128, 256, 0, stream>>>(
        x, W_self_i, neigh, W_neigh_i, b_i, nullptr, nullptr, nullptr, hiso, N_DST_C, 0);

    // ---- 5. base_c = x_dst@W_self_c + b_c (once); h_comm[t] = base_c + mean_t@W_neigh_c ----
    gemm128_kernel<<<(N_DST_C + 127) / 128, 256, 0, stream>>>(
        x, W_self_c, nullptr, nullptr, b_c, nullptr, nullptr, nullptr, hbuf, N_DST_C, 0);
    for (int t = 0; t < 3; ++t) {
        gemm128_kernel<<<(N_DST_C + 127) / 128, 256, 0, stream>>>(
            sum_c + (size_t)t * N_DST_C * 128, W_neigh_c, nullptr, nullptr, nullptr, hbuf,
            nullptr, nullptr, hc + (size_t)t * N_DST_C * 128, N_DST_C, 0);
    }

    // ---- 6. per-tensor unique-row counts, then masked 4-way max -> h (hbuf reused) ----
    hash_count_kernel<<<dim3((N_DST_C + 7) / 8, 4), 256, 0, stream>>>(hiso, hc, tables, uniq);
    combine_max_kernel<<<(N_DST_C * 32 + 255) / 256, 256, 0, stream>>>(hiso, hc, uniq, hbuf);

    // ---- 7. MLP: z1 = relu((h[a]*h[b])@W1 + b1) with fused gather-mul A ----
    gemm128_kernel<<<(NP_C + 127) / 128, 256, 0, stream>>>(
        hbuf, W1, nullptr, nullptr, b1, nullptr, pos_src, pos_dst, z1, NP_C, 1);
    gemm128_kernel<<<(NP_C + 127) / 128, 256, 0, stream>>>(
        hbuf, W1, nullptr, nullptr, b1, nullptr, neg_src, neg_dst,
        z1 + (size_t)NP_C * 128, NP_C, 1);

    // ---- 8. z2 = relu(z1@W2 + b2); out = z2@W3 + b3 ----
    gemm128_kernel<<<(NPQ_C + 127) / 128, 256, 0, stream>>>(
        z1, W2, nullptr, nullptr, b2, nullptr, nullptr, nullptr, z2, NPQ_C, 1);
    final_dot_kernel<<<(NPQ_C + 255) / 256, 256, 0, stream>>>(
        z2, W3, b3, (float*)d_out, NPQ_C);
}
